// DualStreamDTIClassifier_61959198212307
// MI455X (gfx1250) — compile-verified, matrix operand order checked by reading, not run on hardware
//
#include <hip/hip_runtime.h>
#include <hip/hip_bf16.h>
#include <math.h>

typedef __bf16 bf16;
typedef __attribute__((ext_vector_type(16))) __bf16 v16bf;
typedef __attribute__((ext_vector_type(8)))  float  v8f;

#define WMMA_BF16(a, b, c) \
  __builtin_amdgcn_wmma_f32_16x16x32_bf16(false, (a), false, (b), (short)0, (c), false, false)

// ---------------------------------------------------------------------------
// WMMA fragment loaders (wave32).
// A (16x32, MxK): lane holds row r=lane&15; elems [0..7] = K[q*8..q*8+7],
//                 elems [8..15] = K[16+q*8 ..], q = lane>>4.
// B (32x16, KxN): lane holds col n=lane&15 with the same K chunking.
// C/D (16x16):    elem i -> row = i + 8*(lane>>4), col = lane&15.
// ---------------------------------------------------------------------------
__device__ __forceinline__ v16bf frag_a_bf16(const bf16* s, int ld) {
  int lane = threadIdx.x & 31;
  int r = lane & 15, q = lane >> 4;
  const bf16* p0 = s + r * ld + q * 8;
  const bf16* p1 = p0 + 16;
  v16bf a;
#pragma unroll
  for (int i = 0; i < 8; ++i) { a[i] = p0[i]; a[i + 8] = p1[i]; }
  return a;
}

__device__ __forceinline__ v16bf frag_a_f32(const float* s, int ld) {
  int lane = threadIdx.x & 31;
  int r = lane & 15, q = lane >> 4;
  const float* p0 = s + r * ld + q * 8;
  const float* p1 = p0 + 16;
  v16bf a;
#pragma unroll
  for (int i = 0; i < 8; ++i) { a[i] = (bf16)p0[i]; a[i + 8] = (bf16)p1[i]; }
  return a;
}

__device__ __forceinline__ v16bf frag_b_bf16(const bf16* s, int ld) {
  int lane = threadIdx.x & 31;
  int n = lane & 15, q = lane >> 4;
  v16bf b;
#pragma unroll
  for (int i = 0; i < 8; ++i) {
    b[i]     = s[(q * 8 + i) * ld + n];
    b[i + 8] = s[(16 + q * 8 + i) * ld + n];
  }
  return b;
}

// ---------------------------------------------------------------------------
// Weight conversion / packing
// ---------------------------------------------------------------------------
__global__ void f2b_kernel(const float* __restrict__ in, bf16* __restrict__ out, int n) {
  int i = blockIdx.x * 256 + threadIdx.x;
  if (i < n) out[i] = (bf16)in[i];
}

// (H, D, DH) fp32 -> (D, H*DH) bf16 so per-head projection is one GEMM
__global__ void pack_qkv_kernel(const float* __restrict__ in, bf16* __restrict__ out) {
  int i = blockIdx.x * 256 + threadIdx.x;       // over H*D*DH = 1M
  if (i >= 1024 * 1024) return;
  int e = i & 63;
  int d = (i >> 6) & 1023;
  int h = i >> 16;
  out[(size_t)d * 1024 + h * 64 + e] = (bf16)in[i];
}

// ---------------------------------------------------------------------------
// LayerNorm: one row (D=1024) per 256-thread block, bf16 out
// ---------------------------------------------------------------------------
__global__ __launch_bounds__(256) void ln_kernel(
    const float* __restrict__ x, const float* __restrict__ g,
    const float* __restrict__ bta, bf16* __restrict__ out) {
  const int D = 1024;
  int row = blockIdx.x;
  const float* xr = x + (size_t)row * D;
  float s = 0.f, s2 = 0.f;
  for (int i = threadIdx.x; i < D; i += 256) { float v = xr[i]; s += v; s2 += v * v; }
  __shared__ float rs[256], rs2[256];
  rs[threadIdx.x] = s; rs2[threadIdx.x] = s2;
  __syncthreads();
  for (int o = 128; o > 0; o >>= 1) {
    if ((int)threadIdx.x < o) { rs[threadIdx.x] += rs[threadIdx.x + o]; rs2[threadIdx.x] += rs2[threadIdx.x + o]; }
    __syncthreads();
  }
  float mean = rs[0] * (1.f / D);
  float var  = rs2[0] * (1.f / D) - mean * mean;
  float rstd = rsqrtf(var + 1e-5f);
  bf16* orow = out + (size_t)row * D;
  for (int i = threadIdx.x; i < D; i += 256)
    orow[i] = (bf16)((xr[i] - mean) * rstd * g[i] + bta[i]);
}

// ---------------------------------------------------------------------------
// Per-head L2 normalize (F.normalize, eps 1e-12); Q additionally folds
// SCALE/TEMP. Block = 256 threads = one row; 16 threads x 4 elems per head.
// ---------------------------------------------------------------------------
__global__ __launch_bounds__(256) void qknorm_kernel(
    const float* __restrict__ x, bf16* __restrict__ out, float scale) {
  const int D = 1024;
  size_t base = (size_t)blockIdx.x * D + (size_t)threadIdx.x * 4;
  float v[4];
  float ss = 0.f;
#pragma unroll
  for (int i = 0; i < 4; ++i) { v[i] = x[base + i]; ss += v[i] * v[i]; }
#pragma unroll
  for (int m = 1; m < 16; m <<= 1) ss += __shfl_xor(ss, m, 32);
  float nrm = sqrtf(ss);
  float f = scale / fmaxf(nrm, 1e-12f);
#pragma unroll
  for (int i = 0; i < 4; ++i) out[base + i] = (bf16)(v[i] * f);
}

// ---------------------------------------------------------------------------
// Tiled WMMA GEMM: C(MxN) = A(MxK) * B(KxN) [+bias][+resid|+=out][act]
// 256 threads = 8 waves; block tile 128x128; wave w -> rows [16w,16w+16),
// all 128 cols (8 WMMA tiles, acc[8]); K-step 32, LDS-staged A/B with
// global_prefetch of the next K panel (gfx1250 global_prefetch_b8).
// ACT: 0 none, 1 tanh, 2 exact gelu. BF16OUT: bf16 vs f32 out. ACCUM: out += .
// ---------------------------------------------------------------------------
template <int ACT, bool BF16OUT, bool ACCUM>
__global__ __launch_bounds__(256)
void gemm_bf16_kernel(const bf16* __restrict__ A, const bf16* __restrict__ Bw,
                      const float* __restrict__ bias, const float* __restrict__ resid,
                      void* __restrict__ outv, int M, int N, int K) {
  __shared__ bf16 As[128 * 32];
  __shared__ bf16 Bs[32 * 128];
  int m0 = blockIdx.y * 128, n0 = blockIdx.x * 128;
  int wave = threadIdx.x >> 5;
  int tid = threadIdx.x;

  v8f acc[8];
#pragma unroll
  for (int t = 0; t < 8; ++t)
#pragma unroll
    for (int i = 0; i < 8; ++i) acc[t][i] = 0.f;

  for (int k0 = 0; k0 < K; k0 += 32) {
    // stage tiles to LDS
#pragma unroll 4
    for (int i = tid; i < 128 * 32; i += 256) {
      int r = i >> 5, c = i & 31;
      As[i] = A[(size_t)(m0 + r) * K + k0 + c];
    }
#pragma unroll 4
    for (int i = tid; i < 32 * 128; i += 256) {
      int r = i >> 7, c = i & 127;
      Bs[i] = Bw[(size_t)(k0 + r) * N + n0 + c];
    }
    // prefetch next K panel while this step's WMMAs run
    if (k0 + 32 < K) {
      __builtin_prefetch(&A[(size_t)(m0 + (tid >> 1)) * K + k0 + 32], 0, 3);
      __builtin_prefetch(&Bw[(size_t)(k0 + 32 + (tid >> 3)) * N + n0 + ((tid & 7) << 4)], 0, 3);
    }
    __syncthreads();
    v16bf af = frag_a_bf16(&As[wave * 16 * 32], 32);
#pragma unroll
    for (int t = 0; t < 8; ++t) {
      v16bf bfv = frag_b_bf16(&Bs[t * 16], 128);
      acc[t] = WMMA_BF16(af, bfv, acc[t]);
    }
    __syncthreads();
  }

  int lane = threadIdx.x & 31;
  int cn = lane & 15, rh = (lane >> 4) * 8;
#pragma unroll
  for (int t = 0; t < 8; ++t) {
#pragma unroll
    for (int i = 0; i < 8; ++i) {
      int gm = m0 + wave * 16 + rh + i;
      int gn = n0 + t * 16 + cn;
      float v = acc[t][i] + (bias ? bias[gn] : 0.f);
      if (ACT == 1) v = tanhf(v);
      if (ACT == 2) v = 0.5f * v * (1.f + erff(v * 0.70710678118f));
      size_t idx = (size_t)gm * N + gn;
      if (BF16OUT) {
        ((bf16*)outv)[idx] = (bf16)v;
      } else {
        float r0 = ACCUM ? ((float*)outv)[idx] : (resid ? resid[idx] : 0.f);
        ((float*)outv)[idx] = r0 + v;
      }
    }
  }
}

// ---------------------------------------------------------------------------
// Attention: one wave per (b, h, 16-query tile).
// logits via WMMA -> fp32 row in dynamic LDS -> softmax, smoothing, gate,
// exact top-k threshold (binary search on float bit patterns) -> ctx via WMMA.
// dyn LDS = 16 * Lk * 4 bytes.
// ---------------------------------------------------------------------------
__global__ __launch_bounds__(32)
void attn_kernel(const bf16* __restrict__ Q, const bf16* __restrict__ K,
                 const bf16* __restrict__ V, const unsigned char* __restrict__ pad,
                 bf16* __restrict__ ctx, int Lq, int Lk, int ktop) {
  extern __shared__ char smem[];
  float* lg = (float*)smem;                  // [16][Lk]
  const int D = 1024;
  int lane = threadIdx.x;
  int qt = blockIdx.x, h = blockIdx.y, b = blockIdx.z;
  size_t qrow0 = (size_t)b * Lq + (size_t)qt * 16;
  const bf16* Qp = Q + qrow0 * D + h * 64;
  const bf16* Kp = K + (size_t)b * Lk * D + h * 64;
  const bf16* Vp = V + (size_t)b * Lk * D + h * 64;
  const unsigned char* pp = pad + (size_t)b * Lk;

  // ---- logits = (scaled,normalized Q) x K^T ----
  for (int kt = 0; kt < Lk; kt += 16) {
    v8f acc;
#pragma unroll
    for (int i = 0; i < 8; ++i) acc[i] = 0.f;
#pragma unroll
    for (int ks = 0; ks < 2; ++ks) {
      v16bf af = frag_a_bf16(Qp + ks * 32, D);
      v16bf bfk = frag_a_bf16(Kp + (size_t)kt * D + ks * 32, D);  // K^T frag
      acc = WMMA_BF16(af, bfk, acc);
    }
    if (kt + 16 < Lk)
      __builtin_prefetch(Kp + (size_t)(kt + 16 + (lane >> 1)) * D, 0, 3);
    int n = lane & 15, rb = (lane >> 4) * 8;
#pragma unroll
    for (int i = 0; i < 8; ++i) {
      float v = acc[i];
      if (pp[kt + n]) v = -1e4f;
      lg[(rb + i) * Lk + kt + n] = v;
    }
  }
  __syncthreads();

  // ---- per-row softmax + smoothing + gate + top-k ----
  float invLk = 1.0f / (float)Lk;
  for (int r = 0; r < 16; ++r) {
    float* row = lg + r * Lk;
    float m = -3.0e38f;
    for (int j = lane; j < Lk; j += 32) m = fmaxf(m, row[j]);
#pragma unroll
    for (int o = 16; o > 0; o >>= 1) m = fmaxf(m, __shfl_xor(m, o, 32));
    float s = 0.f;
    for (int j = lane; j < Lk; j += 32) { float e = expf(row[j] - m); row[j] = e; s += e; }
#pragma unroll
    for (int o = 16; o > 0; o >>= 1) s += __shfl_xor(s, o, 32);
    float inv = 1.0f / s;
    for (int j = lane; j < Lk; j += 32) {
      float a = row[j] * inv;
      a = 0.95f * a + 0.05f * invLk;        // smoothing
      a = (a >= 0.01f) ? a : 0.0f;          // gate threshold
      row[j] = a;
    }
    // top-k threshold: largest t with count(a >= t) >= ktop == ktop-th largest
    unsigned lo = 0u, hi = 0x7f800000u;
    for (int it = 0; it < 32; ++it) {
      unsigned mid = (lo + hi) >> 1;
      float t = __uint_as_float(mid);
      int cnt = 0;
      for (int j = lane; j < Lk; j += 32) cnt += (row[j] >= t) ? 1 : 0;
#pragma unroll
      for (int o = 16; o > 0; o >>= 1) cnt += __shfl_xor(cnt, o, 32);
      if (cnt >= ktop) lo = mid; else hi = mid;
    }
    float thr = __uint_as_float(lo);
    for (int j = lane; j < Lk; j += 32) {
      float a = row[j];
      row[j] = (a >= thr) ? a : 0.0f;
    }
  }
  __syncthreads();

  // ---- ctx = attn x V ----
  v8f acc[4];
#pragma unroll
  for (int t = 0; t < 4; ++t)
#pragma unroll
    for (int i = 0; i < 8; ++i) acc[t][i] = 0.f;
  for (int k0 = 0; k0 < Lk; k0 += 32) {
    v16bf af = frag_a_f32(lg + k0, Lk);
#pragma unroll
    for (int t = 0; t < 4; ++t) {
      v16bf bfv = frag_b_bf16(Vp + (size_t)k0 * D + t * 16, D);
      acc[t] = WMMA_BF16(af, bfv, acc[t]);
    }
  }
  int n = lane & 15, rb = (lane >> 4) * 8;
#pragma unroll
  for (int t = 0; t < 4; ++t)
#pragma unroll
    for (int i = 0; i < 8; ++i)
      ctx[(qrow0 + rb + i) * D + h * 64 + t * 16 + n] = (bf16)acc[t][i];
}

// ---------------------------------------------------------------------------
// Host: full pipeline
// ---------------------------------------------------------------------------
extern "C" void kernel_launch(void* const* d_in, const int* in_sizes, int n_in,
                              void* d_out, int out_size, void* d_ws, size_t ws_size,
                              hipStream_t stream) {
  const float* p_h  = (const float*)d_in[0];
  const float* l_h  = (const float*)d_in[1];
  const unsigned char* p_pad = (const unsigned char*)d_in[2];
  const unsigned char* l_pad = (const unsigned char*)d_in[3];
  const float* ln_s = (const float*)d_in[4];
  const float* ln_b = (const float*)d_in[5];
  const float* wq = (const float*)d_in[6];
  const float* bq = (const float*)d_in[7];
  const float* wk = (const float*)d_in[8];
  const float* bk = (const float*)d_in[9];
  const float* wv = (const float*)d_in[10];
  const float* bv = (const float*)d_in[11];
  const float* wo = (const float*)d_in[12];
  const float* bo = (const float*)d_in[13];
  const float* w1 = (const float*)d_in[14];
  const float* b1 = (const float*)d_in[15];
  const float* w2 = (const float*)d_in[16];
  const float* b2 = (const float*)d_in[17];
  float* out = (float*)d_out;

  const int B = 8, LP = 1024, LL = 256, D = 1024, H = 16, FF = 4096;
  const int maxRows = B * LP;  // 8192

  size_t off = 0;
  auto alc = [&](size_t bytes) -> char* {
    char* p = (char*)d_ws + off;
    off += (bytes + 255) & ~(size_t)255;
    return p;
  };
  // weights (bf16)
  bf16* wqb = (bf16*)alc((size_t)2 * D * D * 2);
  bf16* wkb = (bf16*)alc((size_t)2 * D * D * 2);
  bf16* wvb = (bf16*)alc((size_t)2 * D * D * 2);
  bf16* wob = (bf16*)alc((size_t)2 * D * D * 2);
  bf16* w1b = (bf16*)alc((size_t)2 * D * FF * 2);
  bf16* w2b = (bf16*)alc((size_t)2 * FF * D * 2);
  // per-stream scratch (sized for worst case, reused across streams)
  bf16*  qln  = (bf16*)alc((size_t)maxRows * D * 2);
  bf16*  kvln = (bf16*)alc((size_t)maxRows * D * 2);
  float* qf   = (float*)alc((size_t)maxRows * D * 4);
  float* kf   = (float*)alc((size_t)maxRows * D * 4);   // contiguous after qf
  bf16*  qb   = (bf16*)alc((size_t)maxRows * D * 2);
  bf16*  kb   = (bf16*)alc((size_t)maxRows * D * 2);
  bf16*  vb   = (bf16*)alc((size_t)maxRows * D * 2);
  bf16*  ctxb = (bf16*)alc((size_t)maxRows * D * 2);
  bf16*  h1b  = (bf16*)qf;    // reuse qf+kf (64MB) for FFN hidden (8192x4096 bf16)
  bf16*  ffln = qln;          // reuse q-LN buffer for FFN layernorm
  if (off > ws_size) return;  // workspace too small; bail deterministically

  // ---- weight packing ----
  {
    int n1 = D * D;
    dim3 g1((n1 + 255) / 256);
    for (int s = 0; s < 2; ++s) {
      pack_qkv_kernel<<<g1, 256, 0, stream>>>(wq + (size_t)s * n1, wqb + (size_t)s * n1);
      pack_qkv_kernel<<<g1, 256, 0, stream>>>(wk + (size_t)s * n1, wkb + (size_t)s * n1);
      pack_qkv_kernel<<<g1, 256, 0, stream>>>(wv + (size_t)s * n1, wvb + (size_t)s * n1);
    }
    int n2 = 2 * D * D;
    f2b_kernel<<<(n2 + 255) / 256, 256, 0, stream>>>(wo, wob, n2);
    int n3 = 2 * D * FF;
    f2b_kernel<<<(n3 + 255) / 256, 256, 0, stream>>>(w1, w1b, n3);
    f2b_kernel<<<(n3 + 255) / 256, 256, 0, stream>>>(w2, w2b, n3);
  }

  for (int s = 0; s < 2; ++s) {
    const float* qsrc = (s == 0) ? l_h : p_h;
    const float* ksrc = (s == 0) ? p_h : l_h;
    int Lq = (s == 0) ? LL : LP;
    int Lk = (s == 0) ? LP : LL;
    int rowsQ = B * Lq, rowsK = B * Lk;
    const unsigned char* pad = (s == 0) ? p_pad : l_pad;
    int lnq = (s == 0) ? 0 : 2, lnkv = (s == 0) ? 3 : 1, lnf = (s == 0) ? 4 : 5;
    float* osec = (s == 0) ? (out + (size_t)B * LP * D) : out;  // (p_out, l_out) concat
    int ktop = Lk / 4;  // ceil(Lk * 0.25) for Lk divisible by 4
    size_t woff = (size_t)s * D * D;

    // LayerNorms -> bf16
    ln_kernel<<<rowsQ, 256, 0, stream>>>(qsrc, ln_s + lnq * D, ln_b + lnq * D, qln);
    ln_kernel<<<rowsK, 256, 0, stream>>>(ksrc, ln_s + lnkv * D, ln_b + lnkv * D, kvln);

    // Projections (128x128 block tiles)
    gemm_bf16_kernel<0, false, false><<<dim3(D / 128, rowsQ / 128), 256, 0, stream>>>(
        qln, wqb + woff, bq + (size_t)s * D, nullptr, qf, rowsQ, D, D);
    gemm_bf16_kernel<0, false, false><<<dim3(D / 128, rowsK / 128), 256, 0, stream>>>(
        kvln, wkb + woff, bk + (size_t)s * D, nullptr, kf, rowsK, D, D);
    gemm_bf16_kernel<1, true, false><<<dim3(D / 128, rowsK / 128), 256, 0, stream>>>(
        kvln, wvb + woff, bv + (size_t)s * D, nullptr, vb, rowsK, D, D);

    // qk-norm (fold SCALE/TEMP = 0.125 into Q)
    qknorm_kernel<<<rowsQ, 256, 0, stream>>>(qf, qb, 0.125f);
    qknorm_kernel<<<rowsK, 256, 0, stream>>>(kf, kb, 1.0f);

    // Attention (softmax + smoothing + gate + top-k inside)
    attn_kernel<<<dim3(Lq / 16, H, B), 32, (size_t)16 * Lk * 4, stream>>>(
        qb, kb, vb, pad, ctxb, Lq, Lk, ktop);

    // Output projection + residual -> h2 (written directly to output section)
    gemm_bf16_kernel<0, false, false><<<dim3(D / 128, rowsQ / 128), 256, 0, stream>>>(
        ctxb, wob + woff, bo + (size_t)s * D, qsrc, osec, rowsQ, D, D);

    // FFN: h2 -> LN -> gelu(xW1+b1) -> out += hW2+b2
    ln_kernel<<<rowsQ, 256, 0, stream>>>(osec, ln_s + lnf * D, ln_b + lnf * D, ffln);
    gemm_bf16_kernel<2, true, false><<<dim3(FF / 128, rowsQ / 128), 256, 0, stream>>>(
        ffln, w1b + (size_t)s * D * FF, b1 + (size_t)s * FF, nullptr, h1b, rowsQ, FF, D);
    gemm_bf16_kernel<0, false, true><<<dim3(D / 128, rowsQ / 128), 256, 0, stream>>>(
        h1b, w2b + (size_t)s * FF * D, b2 + (size_t)s * D, nullptr, osec, rowsQ, D, FF);
  }
}
